// TransformerHEA_68418829025778
// MI455X (gfx1250) — compile-verified, matrix operand order checked by reading, not run on hardware
//
#include <hip/hip_runtime.h>
#include <hip/hip_bf16.h>
#include <math.h>

// ---------------------------------------------------------------------------
// MI455X (gfx1250): compute-bound GEMM workload (~0.86 TFLOP vs ~0.5GB HBM
// traffic @ 23.3 TB/s) -> run everything on v_wmma_f32_16x16x32_bf16.
// Weights are converted f32->bf16 and transposed to [N,K] ONCE per call, so
// GEMM B tiles stream straight into LDS via global_load_async_to_lds_b128
// (ASYNCcnt path). Activations are converted in-flight with packed cvt.
// ---------------------------------------------------------------------------

typedef __attribute__((ext_vector_type(16))) __bf16 bf16x16;
typedef __attribute__((ext_vector_type(8)))  __bf16 bf16x8;
typedef __attribute__((ext_vector_type(4)))  __bf16 bf16x4;
typedef __attribute__((ext_vector_type(8)))  float  f32x8;
typedef __attribute__((ext_vector_type(4)))  float  f32x4;

#define TILE_M 64
#define TILE_N 64
#define TILE_K 64
#define LSTRIDE 72  // padded bf16 stride: 144B rows -> 16B-aligned, conflict-free b128

__device__ __forceinline__ int expert_idx(int z, int task, int S_, int P_) {
    if (task < 0) return z;
    return (z < S_) ? z : (S_ + task * P_ + (z - S_));
}

// W[z][k][n] (f32, row-major) -> WT[z][n][k] (bf16, transposed)
__global__ __launch_bounds__(256)
void convert_transpose_kernel(const float* __restrict__ W, __bf16* __restrict__ WT,
                              int Kd, int Nd, long total)
{
    long idx = (long)blockIdx.x * blockDim.x + threadIdx.x;
    if (idx >= total) return;
    long mk  = (long)Kd * Nd;
    long mat = idx / mk;
    long r   = idx - mat * mk;
    int  kk  = (int)(r / Nd);
    int  nn  = (int)(r - (long)kk * Nd);
    WT[mat * mk + (long)nn * Kd + kk] = (__bf16)W[idx];
}

// Out[z] = act( A[z] @ W[eidx(z)] + bias[eidx(z)] (+ R[z]) )
// A:[M,K] f32; WT: logical [K,N] stored transposed bf16 [N,K]; act:0=none,1=relu,2=gelu
__global__ __launch_bounds__(128)
void wmma_gemm_kernel(const float* __restrict__ A, long Asz,
                      const __bf16* __restrict__ WT, long Wsz,
                      const float* __restrict__ bias, long bsz,
                      const float* __restrict__ R, long Rsz,
                      float* __restrict__ Out, long Osz,
                      int M, int N, int K, int act, int task, int S_, int P_)
{
    __shared__ __bf16 As[TILE_M * LSTRIDE];   // [row][k]
    __shared__ __bf16 Bs[TILE_N * LSTRIDE];   // [col][k] (weights pre-transposed)

    const int z    = blockIdx.z;
    const int eidx = expert_idx(z, task, S_, P_);
    const float*  Az  = A  + (long)z * Asz;
    const __bf16* WTz = WT + (long)eidx * Wsz;
    const float*  bz  = bias + (long)eidx * bsz;
    float* Oz = Out + (long)z * Osz;

    const int m0 = blockIdx.y * TILE_M;
    const int n0 = blockIdx.x * TILE_N;

    const int tid   = threadIdx.x;
    const int lane  = tid & 31;
    const int wave  = tid >> 5;      // 4 waves
    const int wr    = wave >> 1;     // 32-row slab
    const int wc    = wave & 1;      // 32-col slab
    const int khalf = lane >> 4;
    const int l16   = lane & 15;

    f32x8 acc[2][2] = {};

    for (int k0 = 0; k0 < K; k0 += TILE_K) {
        // --- B tile (64 cols x 64 k, bf16): async DMA straight into LDS ---
        // 512 x b128 chunks, 4 per thread
#pragma unroll
        for (int i = 0; i < 4; ++i) {
            int slot = tid + i * 128;
            int nn = slot >> 3, k8 = slot & 7;
            const __bf16* src = WTz + (long)(n0 + nn) * K + k0 + k8 * 8;
            unsigned ldsoff = (unsigned)(uintptr_t)(&Bs[nn * LSTRIDE + k8 * 8]);
            asm volatile("global_load_async_to_lds_b128 %0, %1, off"
                         :: "v"(ldsoff), "v"((unsigned long long)(uintptr_t)src)
                         : "memory");
        }
        // --- A tile (64 rows x 64 k, f32 -> bf16, packed cvt) ---
#pragma unroll
        for (int i = 0; i < 8; ++i) {
            int slot = tid + i * 128;          // 1024 float4 slots
            int r = slot >> 4, c4 = slot & 15;
            f32x4 a4 = *reinterpret_cast<const f32x4*>(&Az[(long)(m0 + r) * K + k0 + c4 * 4]);
            bf16x4 b4;
            b4[0] = (__bf16)a4[0]; b4[1] = (__bf16)a4[1];
            b4[2] = (__bf16)a4[2]; b4[3] = (__bf16)a4[3];
            *reinterpret_cast<bf16x4*>(&As[r * LSTRIDE + c4 * 4]) = b4;
        }
        if (k0 + TILE_K < K) {  // prefetch next A tile (global_prefetch_b8)
            __builtin_prefetch(&Az[(long)(m0 + (tid >> 1)) * K + k0 + TILE_K + (tid & 1) * 32], 0, 1);
        }
        asm volatile("s_wait_asynccnt 0x0" ::: "memory");
        __syncthreads();

        // --- two WMMA K-substeps per staged tile ---
#pragma unroll
        for (int ks = 0; ks < TILE_K; ks += 32) {
            bf16x16 afr[2], bfr[2];
#pragma unroll
            for (int mi = 0; mi < 2; ++mi) {
                int row = wr * 32 + mi * 16 + l16;
                bf16x8 lo = *reinterpret_cast<const bf16x8*>(&As[row * LSTRIDE + ks + khalf * 8]);
                bf16x8 hi = *reinterpret_cast<const bf16x8*>(&As[row * LSTRIDE + ks + 16 + khalf * 8]);
                afr[mi] = __builtin_shufflevector(lo, hi, 0,1,2,3,4,5,6,7,8,9,10,11,12,13,14,15);
            }
#pragma unroll
            for (int ni = 0; ni < 2; ++ni) {
                int col = wc * 32 + ni * 16 + l16;
                bf16x8 lo = *reinterpret_cast<const bf16x8*>(&Bs[col * LSTRIDE + ks + khalf * 8]);
                bf16x8 hi = *reinterpret_cast<const bf16x8*>(&Bs[col * LSTRIDE + ks + 16 + khalf * 8]);
                bfr[ni] = __builtin_shufflevector(lo, hi, 0,1,2,3,4,5,6,7,8,9,10,11,12,13,14,15);
            }
#pragma unroll
            for (int mi = 0; mi < 2; ++mi)
#pragma unroll
                for (int ni = 0; ni < 2; ++ni)
                    acc[mi][ni] = __builtin_amdgcn_wmma_f32_16x16x32_bf16(
                        false, afr[mi], false, bfr[ni],
                        (short)0, acc[mi][ni], false, false);
        }
        __syncthreads();
    }

    // Epilogue: C/D layout -> VGPR r holds row (r + 8*khalf), col = lane%16
#pragma unroll
    for (int mi = 0; mi < 2; ++mi)
#pragma unroll
        for (int ni = 0; ni < 2; ++ni)
#pragma unroll
            for (int r = 0; r < 8; ++r) {
                int gr = m0 + wr * 32 + mi * 16 + khalf * 8 + r;
                int gc = n0 + wc * 32 + ni * 16 + l16;
                float v = acc[mi][ni][r] + bz[gc];
                if (R) v += R[(long)z * Rsz + (long)gr * N + gc];
                if (act == 1)      v = fmaxf(v, 0.0f);
                else if (act == 2) v = 0.5f * v * (1.0f + erff(v * 0.7071067811865475f));
                Oz[(long)gr * N + gc] = v;
            }
}

// Out[z,b,:] = LN(X[z,b,:] + Aadd[z,b,:]) ; H fixed 512, 256 threads x 2 elems
__global__ __launch_bounds__(256)
void addln_kernel(const float* __restrict__ X, const float* __restrict__ Aadd,
                  float* __restrict__ Out,
                  const float* __restrict__ Gm, const float* __restrict__ Bt,
                  long gsz, int H_, int task, int S_, int P_)
{
    const int b = blockIdx.x;
    const int z = blockIdx.y;
    const int eidx = expert_idx(z, task, S_, P_);
    const long rowOff = ((long)z * gridDim.x + b) * H_;
    const float* xr = X + rowOff;
    const float* ar = Aadd ? (Aadd + rowOff) : nullptr;
    float* orow = Out + rowOff;
    const float* g  = Gm + (long)eidx * gsz;
    const float* bt = Bt + (long)eidx * gsz;

    const int tid = threadIdx.x;
    float x0 = xr[tid]       + (ar ? ar[tid]       : 0.0f);
    float x1 = xr[tid + 256] + (ar ? ar[tid + 256] : 0.0f);
    float s = x0 + x1, ss = x0 * x0 + x1 * x1;
#pragma unroll
    for (int off = 16; off; off >>= 1) {
        s  += __shfl_xor(s, off, 32);
        ss += __shfl_xor(ss, off, 32);
    }
    __shared__ float rs[8], rss[8], mv[2];
    const int lane = tid & 31, w = tid >> 5;
    if (lane == 0) { rs[w] = s; rss[w] = ss; }
    __syncthreads();
    if (tid == 0) {
        float S0 = 0.f, S1 = 0.f;
        for (int i = 0; i < 8; ++i) { S0 += rs[i]; S1 += rss[i]; }
        float m = S0 / (float)H_;
        float var = S1 / (float)H_ - m * m;
        mv[0] = m; mv[1] = rsqrtf(var + 1e-5f);
    }
    __syncthreads();
    const float m = mv[0], inv = mv[1];
    orow[tid]       = (x0 - m) * inv * g[tid]       + bt[tid];
    orow[tid + 256] = (x1 - m) * inv * g[tid + 256] + bt[tid + 256];
}

// g = softmax(G1 @ Wg2 + bg2) over NE=6; thread-per-row (tiny)
__global__ void gate2_softmax_kernel(const float* __restrict__ G1,
                                     const float* __restrict__ Wg2,
                                     const float* __restrict__ bg2,
                                     float* __restrict__ Gout,
                                     int B_, int G_, int NE_)
{
    int b = blockIdx.x * blockDim.x + threadIdx.x;
    if (b >= B_) return;
    float s[8];
    float mx = -1e30f;
    for (int n = 0; n < NE_; ++n) {
        float acc = bg2[n];
        for (int k = 0; k < G_; ++k) acc += G1[(long)b * G_ + k] * Wg2[k * NE_ + n];
        s[n] = acc; mx = fmaxf(mx, acc);
    }
    float den = 0.f;
    for (int n = 0; n < NE_; ++n) { s[n] = expf(s[n] - mx); den += s[n]; }
    float r = 1.0f / den;
    for (int n = 0; n < NE_; ++n) Gout[(long)b * NE_ + n] = s[n] * r;
}

// Cat[b, task*D + d] = sum_n G[b,n] * Outs[n,b,d]
__global__ void combine_kernel(const float* __restrict__ Outs,
                               const float* __restrict__ G,
                               float* __restrict__ Cat,
                               int B_, int D_, int NE_, int T_, int task)
{
    int idx = blockIdx.x * blockDim.x + threadIdx.x;
    if (idx >= B_ * D_) return;
    int b = idx / D_, d = idx - b * D_;
    float acc = 0.f;
    for (int n = 0; n < NE_; ++n)
        acc += G[(long)b * NE_ + n] * Outs[((long)n * B_ + b) * D_ + d];
    Cat[(long)b * (T_ * D_) + (long)task * D_ + d] = acc;
}

extern "C" void kernel_launch(void* const* d_in, const int* in_sizes, int n_in,
                              void* d_out, int out_size, void* d_ws, size_t ws_size,
                              hipStream_t stream) {
    const int Dc = 512, Hc = 512, Fc = 2048, Lc = 2;
    const int Sc = 4, Pc = 2, Tc = 3, NEc = Sc + Pc, Ec = Sc + Tc * Pc;  // 6, 10
    const int Gc = 64, Bc = 4096;
    const long BH = (long)Bc * Hc;

    const float* x    = (const float*)d_in[0];
    const float* Win  = (const float*)d_in[1];
    const float* b_in = (const float*)d_in[2];
    const float* Wv   = (const float*)d_in[3];
    const float* bv   = (const float*)d_in[4];
    const float* Wo   = (const float*)d_in[5];
    const float* bo   = (const float*)d_in[6];
    const float* ln1g = (const float*)d_in[7];
    const float* ln1b = (const float*)d_in[8];
    const float* W1   = (const float*)d_in[9];
    const float* b1   = (const float*)d_in[10];
    const float* W2   = (const float*)d_in[11];
    const float* b2   = (const float*)d_in[12];
    const float* ln2g = (const float*)d_in[13];
    const float* ln2b = (const float*)d_in[14];
    const float* lnfg = (const float*)d_in[15];
    const float* lnfb = (const float*)d_in[16];
    const float* Wout = (const float*)d_in[17];
    const float* bout = (const float*)d_in[18];
    const float* Wg1  = (const float*)d_in[19];
    const float* bg1  = (const float*)d_in[20];
    const float* Wg2  = (const float*)d_in[21];
    const float* bg2  = (const float*)d_in[22];
    const float* Wf1  = (const float*)d_in[23];
    const float* bf1  = (const float*)d_in[24];
    const float* Wf2  = (const float*)d_in[25];
    const float* bf2  = (const float*)d_in[26];

    // ---- workspace: f32 activation scratch, then bf16 transposed weights ----
    float* ws  = (float*)d_ws;
    float* h   = ws;                              // [NE,B,H]
    float* v   = h  + (long)NEc * BH;             // [NE,B,H]
    float* aB  = v  + (long)NEc * BH;             // [NE,B,H]
    float* f   = aB + (long)NEc * BH;             // [NE,B,F]
    float* o   = f  + (long)NEc * Bc * Fc;        // [NE,B,D]
    float* cat = o  + (long)NEc * Bc * Dc;        // [B,T*D]
    float* g1  = cat + (long)Bc * Tc * Dc;        // [B,G]
    float* gg  = g1 + (long)Bc * Gc;              // [B,NE]
    float* hid = gg + (long)Bc * NEc;             // [B,2D]

    __bf16* bw   = (__bf16*)(hid + (long)Bc * 2 * Dc);
    __bf16* WinT  = bw;    bw += (long)Ec * Dc * Hc;          // [E][H][D]
    __bf16* WvT   = bw;    bw += (long)Ec * Lc * Hc * Hc;     // [E*L][H][H]
    __bf16* WoT   = bw;    bw += (long)Ec * Lc * Hc * Hc;
    __bf16* W1T   = bw;    bw += (long)Ec * Lc * Hc * Fc;     // [E*L][F][H]
    __bf16* W2T   = bw;    bw += (long)Ec * Lc * Fc * Hc;     // [E*L][H][F]
    __bf16* WoutT = bw;    bw += (long)Ec * Hc * Dc;          // [E][D][H]
    __bf16* Wg1T  = bw;    bw += (long)Tc * Dc * Gc;          // [T][G][D]
    __bf16* Wf1T  = bw;    bw += (long)(Tc * Dc) * (2 * Dc);  // [2D][T*D]
    __bf16* Wf2T  = bw;    bw += (long)(2 * Dc) * Dc;         // [D][2D]

    auto conv = [&](const float* Wsrc, __bf16* Wdst, int Kd, int Nd, long nz) {
        long total = nz * (long)Kd * Nd;
        hipLaunchKernelGGL(convert_transpose_kernel,
                           dim3((unsigned)((total + 255) / 256)), dim3(256), 0, stream,
                           Wsrc, Wdst, Kd, Nd, total);
    };
    // one-time (per call) weight convert+transpose: ~240MB read, ~10us @23TB/s
    conv(Win,  WinT,  Dc,      Hc,     Ec);
    conv(Wv,   WvT,   Hc,      Hc,     (long)Ec * Lc);
    conv(Wo,   WoT,   Hc,      Hc,     (long)Ec * Lc);
    conv(W1,   W1T,   Hc,      Fc,     (long)Ec * Lc);
    conv(W2,   W2T,   Fc,      Hc,     (long)Ec * Lc);
    conv(Wout, WoutT, Hc,      Dc,     Ec);
    conv(Wg1,  Wg1T,  Dc,      Gc,     Tc);
    conv(Wf1,  Wf1T,  Tc * Dc, 2 * Dc, 1);
    conv(Wf2,  Wf2T,  2 * Dc,  Dc,     1);

    auto gemm = [&](const float* A, long Asz, const __bf16* W, long Wsz,
                    const float* bias, long bsz, const float* R, long Rsz,
                    float* Out, long Osz, int M, int N, int K,
                    int act, int task, int Z) {
        dim3 grid(N / TILE_N, M / TILE_M, Z), block(128);
        hipLaunchKernelGGL(wmma_gemm_kernel, grid, block, 0, stream,
                           A, Asz, W, Wsz, bias, bsz, R, Rsz, Out, Osz,
                           M, N, K, act, task, Sc, Pc);
    };
    auto ln = [&](const float* X, const float* Aadd, float* Out,
                  const float* Gm, const float* Bt, long gsz, int task) {
        hipLaunchKernelGGL(addln_kernel, dim3(Bc, NEc), dim3(256), 0, stream,
                           X, Aadd, Out, Gm, Bt, gsz, Hc, task, Sc, Pc);
    };

    for (int t = 0; t < Tc; ++t) {
        const float* xt = x + (long)t * Bc * Dc;

        // gating
        gemm(xt, 0, Wg1T + (long)t * Gc * Dc, 0, bg1 + (long)t * Gc, 0,
             nullptr, 0, g1, 0, Bc, Gc, Dc, /*relu*/1, -1, 1);
        hipLaunchKernelGGL(gate2_softmax_kernel, dim3((Bc + 255) / 256), dim3(256), 0, stream,
                           g1, Wg2 + (long)t * Gc * NEc, bg2 + (long)t * NEc, gg, Bc, Gc, NEc);

        // input projection over NE experts (z-batched, x broadcast)
        gemm(xt, 0, WinT, (long)Hc * Dc, b_in, Hc, nullptr, 0,
             h, BH, Bc, Hc, Dc, 0, t, NEc);

        for (int l = 0; l < Lc; ++l) {
            gemm(h, BH, WvT + (long)l * Hc * Hc, (long)Lc * Hc * Hc,
                 bv + (long)l * Hc, (long)Lc * Hc, nullptr, 0,
                 v, BH, Bc, Hc, Hc, 0, t, NEc);
            gemm(v, BH, WoT + (long)l * Hc * Hc, (long)Lc * Hc * Hc,
                 bo + (long)l * Hc, (long)Lc * Hc, nullptr, 0,
                 aB, BH, Bc, Hc, Hc, 0, t, NEc);
            ln(h, aB, h, ln1g + (long)l * Hc, ln1b + (long)l * Hc, (long)Lc * Hc, t);
            gemm(h, BH, W1T + (long)l * Fc * Hc, (long)Lc * Hc * Fc,
                 b1 + (long)l * Fc, (long)Lc * Fc, nullptr, 0,
                 f, (long)Bc * Fc, Bc, Fc, Hc, /*gelu*/2, t, NEc);
            gemm(f, (long)Bc * Fc, W2T + (long)l * Hc * Fc, (long)Lc * Fc * Hc,
                 b2 + (long)l * Hc, (long)Lc * Hc, nullptr, 0,
                 v, BH, Bc, Hc, Fc, 0, t, NEc);
            ln(h, v, h, ln2g + (long)l * Hc, ln2b + (long)l * Hc, (long)Lc * Hc, t);
        }
        ln(h, nullptr, h, lnfg, lnfb, (long)Hc, t);
        gemm(h, BH, WoutT, (long)Dc * Hc, bout, Dc, xt, 0,
             o, (long)Bc * Dc, Bc, Dc, Hc, 0, t, NEc);

        hipLaunchKernelGGL(combine_kernel, dim3((Bc * Dc + 255) / 256), dim3(256), 0, stream,
                           o, gg, cat, Bc, Dc, NEc, Tc, t);
    }

    // output fusion
    gemm(cat, 0, Wf1T, 0, bf1, 0, nullptr, 0, hid, 0, Bc, 2 * Dc, Tc * Dc, /*relu*/1, -1, 1);
    gemm(hid, 0, Wf2T, 0, bf2, 0, nullptr, 0, (float*)d_out, 0, Bc, Dc, 2 * Dc, 0, -1, 1);
}
